// WaveNetModel_18305150615947
// MI455X (gfx1250) — matrix operand order repaired and checked by reading, not compiled, not run on hardware
//
#include <hip/hip_runtime.h>

// ---------------------------------------------------------------------------
// WaveNet forward, fast-wavenet flat layout: x[c][pos], pos = t*n + b.
// Channel-mixes use v_wmma_f32_16x16x32_f16 (wave32 CDNA5 WMMA).
// Deferred-skip head uses GLOBAL_LOAD_ASYNC_TO_LDS_B128 (ASYNCcnt) to
// double-buffer gated tiles straight into LDS while WMMAs run.
// ---------------------------------------------------------------------------

typedef __attribute__((ext_vector_type(16))) _Float16 v16h;
typedef __attribute__((ext_vector_type(8)))  float    v8f;

#define S_CAP 262144          // capacity (floats) per channel of x buffers
#define TAIL  229376          // final flat length (= 16 * 14336 = out_size)
#define TL    14336
#define NL    40

// Pre-swizzled A-matrix table offsets (in _Float16 elements).
#define AWZ_START 0           // 2 mt * 4 kc * 512
#define AWZ_FG    4096        // 40 * (fg2 * mt2 * tap2) * 512
#define AWZ_RES   167936      // 40 * 2 mt * 512
#define AWZ_SKIP  208896      // 40 * 16 mt * 512
#define AWZ_END1  536576      // 16 mt * 8 kc * 512
#define AWZ_TOTAL 602112

// A-matrix 16x32 f16 layout used here (contiguous K, not the ISA table order): lane holds row m = lane%16;
// element j of the v16h maps to K = j + (lane>=16 ? 16 : 0).
__device__ __host__ inline int krel(int lane, int j) {
  return j + ((lane & 16) ? 16 : 0);
}

__device__ inline v8f wmma_f16(v16h a, v16h b, v8f c) {
  return __builtin_amdgcn_wmma_f32_16x16x32_f16(false, a, false, b,
                                                (short)0, c, false, false);
}

// Async 16-byte global -> LDS transfer (ISA §10.7, tracked by ASYNCcnt).
// LDS operand is the wave-relative LDS byte address: for an LDS location the
// low 32 bits of the generic address are exactly that offset (ISA §10.2).
__device__ inline void async_copy_b128(const _Float16* g, _Float16* l) {
  unsigned lds_off = (unsigned)(uintptr_t)l;
  asm volatile("global_load_async_to_lds_b128 %0, %1, off"
               :: "v"(lds_off), "v"(g) : "memory");
}
__device__ inline void wait_async0() {
  asm volatile("s_wait_asynccnt 0x0" ::: "memory");
}

// ---------------------------------------------------------------------------
// Weight prep: f32 -> f16, pre-swizzled into the per-lane A layout.
// ---------------------------------------------------------------------------
__global__ void prep_weights(const float* __restrict__ sw,   // (32,128,1)
                             const float* __restrict__ fw,   // (40,32,32,2)
                             const float* __restrict__ gw,   // (40,32,32,2)
                             const float* __restrict__ rw,   // (40,32,32,1)
                             const float* __restrict__ kw,   // (40,256,32,1)
                             const float* __restrict__ e1,   // (256,256,1)
                             _Float16* __restrict__ awz) {
  int idx = blockIdx.x * 256 + threadIdx.x;
  const int N0 = 4096, N1 = 163840, N2 = 40960, N3 = 327680, N4 = 65536;
  if (idx >= N0 + N1 + N2 + N3 + N4) return;
  float v;
  if (idx < N0) {                      // start conv A: tile = mt*4 + kc
    int e = idx & 511, tile = idx >> 9;
    int lane = e >> 4, j = e & 15;
    int mt = tile >> 2, kc = tile & 3;
    int m = mt * 16 + (lane & 15);
    int k = kc * 32 + krel(lane, j);
    v = sw[m * 128 + k];
  } else if (idx < N0 + N1) {          // filter/gate: tile=((L*2+fg)*2+mt)*2+tap
    int i1 = idx - N0;
    int e = i1 & 511, tile = i1 >> 9;
    int lane = e >> 4, j = e & 15;
    int layer = tile >> 3, t8 = tile & 7;
    int fg = t8 >> 2, mt = (t8 >> 1) & 1, tap = t8 & 1;
    int m = mt * 16 + (lane & 15);
    int c = krel(lane, j);
    const float* W = fg ? gw : fw;
    v = W[((layer * 32 + m) * 32 + c) * 2 + tap];
  } else if (idx < N0 + N1 + N2) {     // res: tile = layer*2 + mt
    int i2 = idx - N0 - N1;
    int e = i2 & 511, tile = i2 >> 9;
    int lane = e >> 4, j = e & 15;
    int layer = tile >> 1, mt = tile & 1;
    int m = mt * 16 + (lane & 15);
    int c = krel(lane, j);
    v = rw[(layer * 32 + m) * 32 + c];
  } else if (idx < N0 + N1 + N2 + N3) { // skip: tile = layer*16 + mt
    int i3 = idx - N0 - N1 - N2;
    int e = i3 & 511, tile = i3 >> 9;
    int lane = e >> 4, j = e & 15;
    int layer = tile >> 4, mt = tile & 15;
    int m = mt * 16 + (lane & 15);
    int c = krel(lane, j);
    v = kw[(layer * 256 + m) * 32 + c];
  } else {                             // end1: tile = mt*8 + kc
    int i4 = idx - N0 - N1 - N2 - N3;
    int e = i4 & 511, tile = i4 >> 9;
    int lane = e >> 4, j = e & 15;
    int mt = tile >> 3, kc = tile & 7;
    int m = mt * 16 + (lane & 15);
    int k = kc * 32 + krel(lane, j);
    v = e1[m * 256 + k];
  }
  awz[idx] = (_Float16)v;
}

__global__ void zero_kernel(float* __restrict__ p, long n) {
  long i = (long)blockIdx.x * blockDim.x + threadIdx.x;
  long stride = (long)gridDim.x * blockDim.x;
  for (; i < n; i += stride) p[i] = 0.f;
}

// ---------------------------------------------------------------------------
// Start conv: x0[32][pos] = W(32x128) * in, pos = t*16 + b.
// ---------------------------------------------------------------------------
__global__ __launch_bounds__(256) void start_conv_kernel(
    const float* __restrict__ in, const _Float16* __restrict__ awz,
    float* __restrict__ x0) {
  int w = threadIdx.x >> 5, lane = threadIdx.x & 31;
  int t = blockIdx.x * 8 + w;
  int nn = lane & 15;
  int kb = (lane & 16) ? 16 : 0;
  int rofs = (lane & 16) ? 8 : 0;
  const _Float16* A = awz + AWZ_START;
  v8f acc0 = {}, acc1 = {};
  for (int kc = 0; kc < 4; ++kc) {
    v16h b;
    for (int j = 0; j < 16; ++j) {
      int c = kc * 32 + kb + j;             // B element (K=c, N=b) = in[b][c][t]
      b[j] = (_Float16)in[((long)nn * 128 + c) * 16384 + t];
    }
    v16h a0 = *(const v16h*)(A + ((0 * 4 + kc) * 32 + lane) * 16);
    v16h a1 = *(const v16h*)(A + ((1 * 4 + kc) * 32 + lane) * 16);
    acc0 = wmma_f16(a0, b, acc0);
    acc1 = wmma_f16(a1, b, acc1);
  }
  long pos = (long)t * 16 + nn;
  for (int r = 0; r < 8; ++r) {
    x0[(size_t)(r + rofs) * S_CAP + pos]      = acc0[r];
    x0[(size_t)(16 + r + rofs) * S_CAP + pos] = acc1[r];
  }
}

// ---------------------------------------------------------------------------
// Fused layer: gated conv (WMMA) -> tanh*sigmoid -> LDS -> res conv (+res add)
// Tail handling: deferred -> archive gated f16; fallback -> skip f32 RMW GEMM.
// ---------------------------------------------------------------------------
__global__ __launch_bounds__(256) void layer_kernel(
    const float* __restrict__ xin, float* __restrict__ xout,
    float* __restrict__ skip, _Float16* __restrict__ garchL,
    const _Float16* __restrict__ awz,
    int layer, int nconv, int padE, long s_out, int deferred) {
  __shared__ __align__(32) _Float16 g_lds[4 * 16 * 32];   // [tile][n][c]
  int w = threadIdx.x >> 5, lane = threadIdx.x & 31;
  int nn = lane & 15;
  int kb = (lane & 16) ? 16 : 0;
  int rofs = (lane & 16) ? 8 : 0;
  long p0b = (long)blockIdx.x * 64;
  long tv = (s_out - p0b + 15) / 16;
  int tilesValid = tv > 4 ? 4 : (int)tv;
  long qbase = s_out - TAIL;

  const _Float16* Afg  = awz + AWZ_FG  + (size_t)layer * 8 * 512;
  const _Float16* Ares = awz + AWZ_RES + (size_t)layer * 2 * 512;
  const _Float16* Askp = awz + AWZ_SKIP + (size_t)layer * 16 * 512;

  // Phase 1: f = Wf*[x(pos); x(pos+n)], g = Wg*[...]; gated = tanh(f)*sig(g)
  int tp = w >> 1, h = w & 1;                      // wave -> (tile, chan-half)
  if (tp < tilesValid) {
    long p0 = p0b + tp * 16;
    v8f f = {}, g = {};
    for (int tap = 0; tap < 2; ++tap) {            // K chunk per conv tap
      v16h b;
      long idx = p0 + nn + (long)tap * nconv - padE;
      for (int j = 0; j < 16; ++j) {
        int c = kb + j;
        float fv = (idx >= 0) ? xin[(size_t)c * S_CAP + idx] : 0.f;
        b[j] = (_Float16)fv;
      }
      v16h af = *(const v16h*)(Afg + (((0 * 2 + h) * 2 + tap) * 32 + lane) * 16);
      v16h ag = *(const v16h*)(Afg + (((1 * 2 + h) * 2 + tap) * 32 + lane) * 16);
      f = wmma_f16(af, b, f);
      g = wmma_f16(ag, b, g);
    }
    int cbase = h * 16 + rofs;
    int inTail = deferred && (p0 >= qbase);        // wave-uniform
    long q = p0 - qbase + nn;
    for (int r = 0; r < 8; ++r) {
      float gv = tanhf(f[r]) * (1.f / (1.f + __expf(-g[r])));
      _Float16 gh = (_Float16)gv;
      g_lds[(tp * 16 + nn) * 32 + cbase + r] = gh;
      if (inTail)                                  // archive for deferred skip
        garchL[(size_t)(cbase + r) * TAIL + q] = gh;
    }
  }
  __syncthreads();

  // Phase 2: res conv + residual add: xout[pos] = Wr*gated[pos] + xpad[pos+n]
  if (tp < tilesValid) {
    long p0 = p0b + tp * 16;
    v16h b = *(const v16h*)(&g_lds[(tp * 16 + nn) * 32 + kb]);
    v16h a = *(const v16h*)(Ares + (h * 32 + lane) * 16);
    v8f d = {};
    d = wmma_f16(a, b, d);
    long pos = p0 + nn;
    for (int r = 0; r < 8; ++r) {
      int m = h * 16 + rofs + r;
      float resv = xin[(size_t)m * S_CAP + pos + nconv - padE];
      xout[(size_t)m * S_CAP + pos] = d[r] + resv;
    }
  }

  // Phase 3 (fallback only): skip conv, right-aligned f32 accumulate.
  if (!deferred) {
    for (int u = w; u < 64; u += 8) {               // (tile, mtile) pairs
      int tpp = u >> 4, mt = u & 15;
      if (tpp >= tilesValid) continue;              // wave-uniform
      long p0 = p0b + tpp * 16;
      long q0 = p0 - qbase;
      if (q0 < 0) continue;                         // wave-uniform
      v16h b = *(const v16h*)(&g_lds[(tpp * 16 + nn) * 32 + kb]);
      v16h a = *(const v16h*)(Askp + (mt * 32 + lane) * 16);
      v8f c;
      long q = q0 + nn;
      for (int r = 0; r < 8; ++r)
        c[r] = skip[(size_t)(mt * 16 + rofs + r) * TAIL + q];
      c = wmma_f16(a, b, c);
      for (int r = 0; r < 8; ++r)
        skip[(size_t)(mt * 16 + rofs + r) * TAIL + q] = c[r];
    }
  }
}

// ---------------------------------------------------------------------------
// Deferred head: skip[m][q] = sum_i Wskip_i * gated_i  (in registers),
// then relu -> end1 (WMMA) -> +bias, relu -> end2 dot -> out.
// Block = 64 positions; gated tiles streamed with async global->LDS b128
// transfers (ASYNCcnt), double-buffered and overlapped with the WMMAs.
// ---------------------------------------------------------------------------
__global__ __launch_bounds__(256) void final_deferred_kernel(
    const _Float16* __restrict__ garch, const _Float16* __restrict__ awz,
    const float* __restrict__ e1b, const float* __restrict__ e2w,
    const float* __restrict__ e2b, float* __restrict__ out) {
  __shared__ __align__(32) _Float16 sg[2][32 * 64];      // gated [c][p]
  __shared__ __align__(32) _Float16 sb[64 * 256];        // relu(skip) [p][c]
  __shared__ float red[4 * 256];
  int w = threadIdx.x >> 5, lane = threadIdx.x & 31;
  int nn = lane & 15;
  int kb = (lane & 16) ? 16 : 0;
  int rofs = (lane & 16) ? 8 : 0;
  long q0 = (long)blockIdx.x * 64;

  // Each thread owns one 16B chunk of the 32x64-half tile: row c, chunk chk.
  int cth = threadIdx.x >> 3, chk = threadIdx.x & 7;
  long goff = (size_t)cth * TAIL + q0 + chk * 8;
  int  loff = cth * 64 + chk * 8;

  // Prime the pipeline with layer 0.
  async_copy_b128(garch + goff, &sg[0][loff]);

  v8f acc[4][2] = {};                                    // [nt][mt-half]
  for (int l = 0; l < NL; ++l) {
    wait_async0();                                       // s_wait_asynccnt 0
    __syncthreads();                                     // tile l visible
    if (l + 1 < NL)                                      // overlap DMA of l+1
      async_copy_b128(garch + (size_t)(l + 1) * 32 * TAIL + goff,
                      &sg[(l + 1) & 1][loff]);
    const _Float16* A = awz + AWZ_SKIP + (size_t)l * 16 * 512;
    v16h a0 = *(const v16h*)(A + ((w)     * 32 + lane) * 16);
    v16h a1 = *(const v16h*)(A + ((w + 8) * 32 + lane) * 16);
    const _Float16* sgb = &sg[l & 1][0];
    for (int nt = 0; nt < 4; ++nt) {
      v16h b;
      for (int j = 0; j < 16; ++j)                       // ds_load_u16 gather
        b[j] = sgb[(kb + j) * 64 + nt * 16 + nn];
      acc[nt][0] = wmma_f16(a0, b, acc[nt][0]);
      acc[nt][1] = wmma_f16(a1, b, acc[nt][1]);
    }
    __syncthreads();                                     // done reading buf l
  }

  // relu(skip) -> LDS [p][c] f16 for the end1 GEMM.
  for (int half = 0; half < 2; ++half) {
    int mt = w + half * 8;
    for (int nt = 0; nt < 4; ++nt)
      for (int r = 0; r < 8; ++r) {
        int m = mt * 16 + rofs + r;
        int p = nt * 16 + nn;
        sb[p * 256 + m] = (_Float16)fmaxf(acc[nt][half][r], 0.f);
      }
  }
  __syncthreads();

  const _Float16* A1 = awz + AWZ_END1;
  float part[4] = {0.f, 0.f, 0.f, 0.f};
  for (int half = 0; half < 2; ++half) {
    int mt = w + half * 8;
    v8f h[4] = {};
    for (int kc = 0; kc < 8; ++kc) {
      v16h a = *(const v16h*)(A1 + ((mt * 8 + kc) * 32 + lane) * 16);
      for (int nt = 0; nt < 4; ++nt) {
        v16h b = *(const v16h*)(&sb[(nt * 16 + nn) * 256 + kc * 32 + kb]);
        h[nt] = wmma_f16(a, b, h[nt]);
      }
    }
    for (int nt = 0; nt < 4; ++nt)
      for (int r = 0; r < 8; ++r) {
        int m = mt * 16 + rofs + r;
        float hv = fmaxf(h[nt][r] + e1b[m], 0.f);
        part[nt] += hv * e2w[m];
      }
  }
  for (int nt = 0; nt < 4; ++nt) red[nt * 256 + threadIdx.x] = part[nt];
  __syncthreads();
  if (threadIdx.x < 64) {
    int nt = threadIdx.x >> 4, n = threadIdx.x & 15;
    float sum = e2b[0];
    for (int ww = 0; ww < 8; ++ww)
      sum += red[nt * 256 + ww * 32 + n] + red[nt * 256 + ww * 32 + 16 + n];
    long t = (long)blockIdx.x * 4 + nt;                  // q = t*16 + b
    out[(size_t)n * TL + t] = sum;                       // (16,1,14336) NCL
  }
}

// ---------------------------------------------------------------------------
// Fallback head (f32 skip accumulator in HBM).
// ---------------------------------------------------------------------------
__global__ __launch_bounds__(256) void final_kernel(
    const float* __restrict__ skip, const _Float16* __restrict__ awz,
    const float* __restrict__ e1b, const float* __restrict__ e2w,
    const float* __restrict__ e2b, float* __restrict__ out) {
  __shared__ __align__(32) _Float16 sb[16 * 256];
  __shared__ float red[256];
  int w = threadIdx.x >> 5, lane = threadIdx.x & 31;
  int nn = lane & 15;
  int kb = (lane & 16) ? 16 : 0;
  int rofs = (lane & 16) ? 8 : 0;
  long q0 = (long)blockIdx.x * 16;

  for (int e = threadIdx.x; e < 4096; e += 256) {
    int c = e >> 4, n = e & 15;
    float v = skip[(size_t)c * TAIL + q0 + n];
    sb[n * 256 + c] = (_Float16)fmaxf(v, 0.f);
  }
  __syncthreads();

  const _Float16* A1 = awz + AWZ_END1;
  float part = 0.f;
  for (int half = 0; half < 2; ++half) {
    int mt = w + half * 8;
    v8f hacc = {};
    for (int kc = 0; kc < 8; ++kc) {
      v16h b = *(const v16h*)(&sb[nn * 256 + kc * 32 + kb]);
      v16h a = *(const v16h*)(A1 + ((mt * 8 + kc) * 32 + lane) * 16);
      hacc = wmma_f16(a, b, hacc);
    }
    for (int r = 0; r < 8; ++r) {
      int m = mt * 16 + rofs + r;
      float hv = fmaxf(hacc[r] + e1b[m], 0.f);
      part += hv * e2w[m];
    }
  }
  red[threadIdx.x] = part;
  __syncthreads();
  if (threadIdx.x < 16) {
    float sum = e2b[0];
    for (int ww = 0; ww < 8; ++ww)
      sum += red[ww * 32 + threadIdx.x] + red[ww * 32 + 16 + threadIdx.x];
    out[(size_t)threadIdx.x * TL + blockIdx.x] = sum;
  }
}

// ---------------------------------------------------------------------------
extern "C" void kernel_launch(void* const* d_in, const int* in_sizes, int n_in,
                              void* d_out, int out_size, void* d_ws,
                              size_t ws_size, hipStream_t stream) {
  const float* input    = (const float*)d_in[0];
  const float* start_w  = (const float*)d_in[1];
  const float* filter_w = (const float*)d_in[2];
  const float* gate_w   = (const float*)d_in[3];
  const float* res_w    = (const float*)d_in[4];
  const float* skip_w   = (const float*)d_in[5];
  const float* end1_w   = (const float*)d_in[6];
  const float* end1_b   = (const float*)d_in[7];
  const float* end2_w   = (const float*)d_in[8];
  const float* end2_b   = (const float*)d_in[9];
  (void)in_sizes; (void)n_in; (void)out_size;

  const size_t xfloats = (size_t)32 * S_CAP;             // per x buffer
  const size_t need_def = xfloats * 2 * 4 + (size_t)NL * 32 * TAIL * 2
                        + (size_t)AWZ_TOTAL * 2;         // ~655 MB
  int deferred = (ws_size >= need_def) ? 1 : 0;

  float* x0 = (float*)d_ws;
  float* x1 = x0 + xfloats;
  float* skip = nullptr;
  _Float16* garch = nullptr;
  _Float16* awz;
  if (deferred) {
    garch = (_Float16*)(x1 + xfloats);                   // 40*32*TAIL halfs
    awz   = garch + (size_t)NL * 32 * TAIL;
  } else {
    skip = x1 + xfloats;                                 // 256*TAIL f32
    awz  = (_Float16*)(skip + (size_t)256 * TAIL);
  }

  prep_weights<<<(AWZ_TOTAL + 255) / 256, 256, 0, stream>>>(
      start_w, filter_w, gate_w, res_w, skip_w, end1_w, awz);
  if (!deferred)
    zero_kernel<<<2048, 256, 0, stream>>>(skip, (long)256 * TAIL);
  start_conv_kernel<<<2048, 256, 0, stream>>>(input, awz, x0);

  // Dilation schedule (mirror of _dilations()).
  int dil[NL], ini[NL];
  { int idx = 0, init = 1;
    for (int b = 0; b < 4; ++b) { int nw = 1;
      for (int l = 0; l < 10; ++l) { dil[idx] = nw; ini[idx] = init;
        init = nw; nw *= 2; ++idx; } } }

  // Flat-space shape evolution, mirroring the Python dilate()/pad()/conv.
  long s = (long)S_CAP;
  int  n = 16;
  float* xin = x0; float* xout = x1;
  for (int i = 0; i < NL; ++i) {
    int d = dil[i], init = ini[i], padE = 0;
    if (d > init) {
      int df = d / init;
      long l = s / n;
      if (l % df) { padE = n; s += n; }   // ConstantPad1d((1,0)) pre-reshape
      n *= df;
    } else if (d < init) {
      n /= (init / d);
    }
    long s_out = s - n;                   // VALID K=2 conv shrinks by n
    int blocks = (int)((s_out + 63) / 64);
    _Float16* gL = deferred ? (garch + (size_t)i * 32 * TAIL) : nullptr;
    layer_kernel<<<blocks, 256, 0, stream>>>(xin, xout, skip, gL, awz,
                                             i, n, padE, s_out, deferred);
    s = s_out;
    float* t = xin; xin = xout; xout = t;
  }
  // After 40 layers s == TAIL (229376) by construction.
  if (deferred)
    final_deferred_kernel<<<TAIL / 64, 256, 0, stream>>>(
        garch, awz, end1_b, end2_w, end2_b, (float*)d_out);
  else
    final_kernel<<<TAIL / 16, 256, 0, stream>>>(
        skip, awz, end1_b, end2_w, end2_b, (float*)d_out);
}
